// FairLSTM_35347580846760
// MI455X (gfx1250) — 8-bit matrix operand order repaired and checked by reading, compiled, not run on hardware
//
#include <hip/hip_runtime.h>

typedef __attribute__((ext_vector_type(8)))  int   v8i;
typedef __attribute__((ext_vector_type(16))) int   v16i;
typedef __attribute__((ext_vector_type(8)))  float v8f;

#define HID      128
#define INPD     64
#define SEQ      128
#define MTILE    32
#define THREADS  256

#define W0_BYTES   (32 * 3 * 1024)
#define W1_BYTES   (32 * 4 * 1024)
#define WD_BYTES   (4 * 2 * 1024)
#define OFF_W0     0
#define OFF_W1     (OFF_W0 + W0_BYTES)
#define OFF_WD     (OFF_W1 + W1_BYTES)
#define WEIGHT_BYTES (OFF_WD + WD_BYTES)
#define OFF_A0     WEIGHT_BYTES
#define OFF_A1     (OFF_A0 + 2 * 3072)
#define LDS_BYTES  (OFF_A1 + 2 * 4096)
#define WS_B0      WEIGHT_BYTES
#define WS_B1      (WS_B0 + 512 * 4)
#define WS_BD      (WS_B1 + 512 * 4)

__device__ inline unsigned int f32_to_fp8_sw(float x) {
  unsigned int u = __float_as_uint(x);
  unsigned int sign = (u >> 31) << 7;
  float ax = __uint_as_float(u & 0x7FFFFFFFu);
  if (!(ax < 464.0f)) ax = 448.0f;
  unsigned int bits = __float_as_uint(ax);
  int e = (int)((bits >> 23) & 0xFF) - 127;
  unsigned int res;
  if (e < -6) {
    int q = (int)rintf(ax * 512.0f);
    res = (unsigned int)q;
  } else {
    unsigned int mant = bits & 0x7FFFFF;
    unsigned int r = mant + 0x7FFFF + ((mant >> 20) & 1);
    if (r & 0x800000) { r = 0; e += 1; }
    if (e > 8) res = 0x7E;
    else       res = (unsigned int)(((e + 7) << 3) | ((r >> 20) & 7));
  }
  return sign | res;
}

__device__ inline unsigned int cvt2_fp8(float a, float b) {
  return (unsigned int)__builtin_amdgcn_cvt_pk_fp8_f32(a, b, 0, false) & 0xFFFFu;
}

__device__ inline void decode_b(int o, int& k_local, int& n_local) {
  int second = o >> 9;
  int o2     = o & 511;
  int lane   = o2 >> 4;
  int rem    = o2 & 15;
  int v      = (second << 2) | (rem >> 2);
  int b      = rem & 3;
  k_local = ((v >> 2) << 5) | ((v & 3) << 2) | ((lane >> 4) << 4) | b;
  n_local = lane & 15;
}

__device__ inline int a_off(int m, int kl) {
  int v_hi   = kl >> 5;
  int within = kl & 31;
  int half   = within >> 4;
  int rem    = within & 15;
  int v      = (v_hi << 2) | (rem >> 2);
  int b      = rem & 3;
  int lane   = (half << 4) + m;
  return ((v >= 4) ? 512 : 0) + lane * 16 + ((v & 3) << 2) + b;
}

__global__ void lstm_prep_weights(const float* __restrict__ w_ih0, const float* __restrict__ w_hh0,
                                  const float* __restrict__ w_ih1, const float* __restrict__ w_hh1,
                                  const float* __restrict__ w_dec, unsigned char* __restrict__ wsb) {
  int idx = blockIdx.x * THREADS + threadIdx.x;
  if (idx >= WEIGHT_BYTES) return;
  float val;
  if (idx < OFF_W1) {
    int t = idx >> 10, o = idx & 1023;
    int nt = t / 3, kt = t % 3, kl, nl;
    decode_b(o, kl, nl);
    int k = kt * 64 + kl, n = nt * 16 + nl;
    val = (k < INPD) ? w_ih0[n * INPD + k] : w_hh0[n * HID + (k - INPD)];
  } else if (idx < OFF_WD) {
    int rel = idx - OFF_W1;
    int t = rel >> 10, o = rel & 1023;
    int nt = t >> 2, kt = t & 3, kl, nl;
    decode_b(o, kl, nl);
    int k = kt * 64 + kl, n = nt * 16 + nl;
    val = (k < HID) ? w_ih1[n * HID + k] : w_hh1[n * HID + (k - HID)];
  } else {
    int rel = idx - OFF_WD;
    int t = rel >> 10, o = rel & 1023;
    int nt = t >> 1, kt = t & 1, kl, nl;
    decode_b(o, kl, nl);
    int k = kt * 64 + kl, n = nt * 16 + nl;
    val = w_dec[n * HID + k];
  }
  wsb[idx] = (unsigned char)f32_to_fp8_sw(val);
}

__global__ void lstm_prep_bias(const float* __restrict__ b_ih0, const float* __restrict__ b_hh0,
                               const float* __restrict__ b_ih1, const float* __restrict__ b_hh1,
                               const float* __restrict__ b_dec, unsigned char* __restrict__ wsb) {
  int i = blockIdx.x * THREADS + threadIdx.x;
  float* b0 = (float*)(wsb + WS_B0);
  float* b1 = (float*)(wsb + WS_B1);
  float* bd = (float*)(wsb + WS_BD);
  if (i < 512)       b0[i] = b_ih0[i] + b_hh0[i];
  else if (i < 1024) b1[i - 512] = b_ih1[i - 512] + b_hh1[i - 512];
  else if (i < 1088) bd[i - 1024] = b_dec[i - 1024];
}

__device__ inline float sigf(float v)  { return __builtin_amdgcn_rcpf(1.0f + __expf(-v)); }
__device__ inline float tanhx(float v) { return __builtin_amdgcn_tanhf(v); }

__device__ inline v8i ld_tile64(const unsigned char* p, int lane) {
  uint4 lo = *(const uint4*)(p + lane * 16);
  uint4 hi = *(const uint4*)(p + 512 + lane * 16);
  v8i r;
  r[0] = lo.x; r[1] = lo.y; r[2] = lo.z; r[3] = lo.w;
  r[4] = hi.x; r[5] = hi.y; r[6] = hi.z; r[7] = hi.w;
  return r;
}

__device__ inline v16i ld_tile128(const unsigned char* p, int lane) {
  uint4 q0 = *(const uint4*)(p + lane * 16);
  uint4 q1 = *(const uint4*)(p + 512 + lane * 16);
  uint4 q2 = *(const uint4*)(p + 1024 + lane * 16);
  uint4 q3 = *(const uint4*)(p + 1536 + lane * 16);
  v16i r;
  r[0] = q0.x;  r[1] = q0.y;  r[2] = q0.z;  r[3] = q0.w;
  r[4] = q1.x;  r[5] = q1.y;  r[6] = q1.z;  r[7] = q1.w;
  r[8] = q2.x;  r[9] = q2.y;  r[10] = q2.z; r[11] = q2.w;
  r[12] = q3.x; r[13] = q3.y; r[14] = q3.z; r[15] = q3.w;
  return r;
}

__device__ inline void nonlin_store(v8f acc[4][2], int mt, v8f& cst, unsigned char* sm,
                                    int j, int halfsel, int dstA, int dstB) {
  int koff = (j >> 6) * 1024;
  int kl   = j & 63;
  float hv[8];
#pragma unroll
  for (int r = 0; r < 8; ++r) {
    float ig = sigf(acc[0][mt][r]);
    float fg = sigf(acc[1][mt][r]);
    float gg = tanhx(acc[2][mt][r]);
    float og = sigf(acc[3][mt][r]);
    float c  = fg * cst[r] + ig * gg;
    cst[r] = c;
    hv[r] = og * tanhx(c);
  }
#pragma unroll
  for (int r = 0; r < 8; r += 2) {
    unsigned int u = cvt2_fp8(hv[r], hv[r + 1]);
    int o0 = koff + a_off(r + (halfsel << 3), kl);
    int o1 = koff + a_off(r + 1 + (halfsel << 3), kl);
    sm[dstA + o0] = (unsigned char)(u & 0xFF);
    sm[dstA + o1] = (unsigned char)(u >> 8);
    if (dstB >= 0) {
      sm[dstB + o0] = (unsigned char)(u & 0xFF);
      sm[dstB + o1] = (unsigned char)(u >> 8);
    }
  }
}

__global__ __launch_bounds__(THREADS, 1)
void lstm_main(const float* __restrict__ x, const unsigned char* __restrict__ ws,
               float* __restrict__ out, int pred_steps) {
  extern __shared__ unsigned char smem[];
  const int tid     = threadIdx.x;
  const int wv      = tid >> 5;
  const int lane    = tid & 31;
  const int nl      = lane & 15;
  const int halfsel = lane >> 4;
  const int m0      = blockIdx.x * MTILE;
  const int jcol    = wv * 16 + nl;

  {
    const uint4* src = (const uint4*)ws;
    uint4* dst = (uint4*)smem;
    for (int i = tid; i < WEIGHT_BYTES / 16; i += THREADS) dst[i] = src[i];
    for (int i = tid; i < 2 * 3072 + 2 * 4096; i += THREADS) smem[OFF_A0 + i] = 0;
  }
  const float* b0p = (const float*)(ws + WS_B0);
  const float* b1p = (const float*)(ws + WS_B1);
  const float* bdp = (const float*)(ws + WS_BD);
  float bias0[4], bias1[4];
#pragma unroll
  for (int g = 0; g < 4; ++g) {
    bias0[g] = b0p[g * HID + jcol];
    bias1[g] = b1p[g * HID + jcol];
  }
  float biasd = (wv < 4) ? bdp[jcol] : 0.0f;

  v8f c0[2] = {{}, {}}, c1[2] = {{}, {}};
  __syncthreads();

  const int nsteps = SEQ + pred_steps;
  for (int t = 0; t < nsteps; ++t) {
    const bool enc = (t < SEQ);
    if (enc) {
#pragma unroll
      for (int hh = 0; hh < 2; ++hh) {
        int e = tid * 4 + hh * 1024;
        int m = e >> 6, k = e & 63;
        const float4 xv = *(const float4*)(x + ((size_t)(m0 + m) * SEQ + t) * INPD + k);
        unsigned int pk = cvt2_fp8(xv.x, xv.y) | (cvt2_fp8(xv.z, xv.w) << 16);
        *(unsigned int*)(smem + OFF_A0 + (m >> 4) * 3072 + a_off(m & 15, k)) = pk;
      }
    }
    __syncthreads();

    v8f acc[4][2];
#pragma unroll
    for (int g = 0; g < 4; ++g)
#pragma unroll
      for (int mt = 0; mt < 2; ++mt) {
        v8f a;
#pragma unroll
        for (int i = 0; i < 8; ++i) a[i] = bias0[g];
        acc[g][mt] = a;
      }
    {
      v16i A128[2]; v8i A64[2];
#pragma unroll
      for (int mt = 0; mt < 2; ++mt) {
        A128[mt] = ld_tile128(smem + OFF_A0 + mt * 3072, lane);
        A64[mt]  = ld_tile64(smem + OFF_A0 + mt * 3072 + 2048, lane);
      }
#pragma unroll
      for (int g = 0; g < 4; ++g) {
        const unsigned char* wp = smem + OFF_W0 + (g * 8 + wv) * 3072;
        v16i B128 = ld_tile128(wp, lane);
        v8i  B64  = ld_tile64(wp + 2048, lane);
#pragma unroll
        for (int mt = 0; mt < 2; ++mt) {
          acc[g][mt] = __builtin_amdgcn_wmma_f32_16x16x128_fp8_fp8(A128[mt], B128, (short)0, acc[g][mt], false, false);
          acc[g][mt] = __builtin_amdgcn_wmma_f32_16x16x64_fp8_fp8(A64[mt], B64, (short)0, acc[g][mt], false, false);
        }
      }
    }
    __syncthreads();
#pragma unroll
    for (int mt = 0; mt < 2; ++mt)
      nonlin_store(acc, mt, c0[mt], smem, jcol, halfsel,
                   OFF_A1 + mt * 4096, OFF_A0 + mt * 3072 + 1024);
    __syncthreads();

#pragma unroll
    for (int g = 0; g < 4; ++g)
#pragma unroll
      for (int mt = 0; mt < 2; ++mt) {
        v8f a;
#pragma unroll
        for (int i = 0; i < 8; ++i) a[i] = bias1[g];
        acc[g][mt] = a;
      }
    {
      v16i Aa[2], Ab[2];
#pragma unroll
      for (int mt = 0; mt < 2; ++mt) {
        Aa[mt] = ld_tile128(smem + OFF_A1 + mt * 4096, lane);
        Ab[mt] = ld_tile128(smem + OFF_A1 + mt * 4096 + 2048, lane);
      }
#pragma unroll
      for (int g = 0; g < 4; ++g) {
        const unsigned char* wp = smem + OFF_W1 + (g * 8 + wv) * 4096;
        v16i B1 = ld_tile128(wp, lane);
        v16i B2 = ld_tile128(wp + 2048, lane);
#pragma unroll
        for (int mt = 0; mt < 2; ++mt) {
          acc[g][mt] = __builtin_amdgcn_wmma_f32_16x16x128_fp8_fp8(Aa[mt], B1, (short)0, acc[g][mt], false, false);
          acc[g][mt] = __builtin_amdgcn_wmma_f32_16x16x128_fp8_fp8(Ab[mt], B2, (short)0, acc[g][mt], false, false);
        }
      }
    }
    __syncthreads();
#pragma unroll
    for (int mt = 0; mt < 2; ++mt)
      nonlin_store(acc, mt, c1[mt], smem, jcol, halfsel,
                   OFF_A1 + mt * 4096 + 2048, -1);

    if (!enc) {
      __syncthreads();
      const int d = t - SEQ;
      if (wv < 4) {
        v8f pa[2];
#pragma unroll
        for (int mt = 0; mt < 2; ++mt) {
#pragma unroll
          for (int i = 0; i < 8; ++i) pa[mt][i] = biasd;
        }
        v16i B = ld_tile128(smem + OFF_WD + wv * 2048, lane);
#pragma unroll
        for (int mt = 0; mt < 2; ++mt) {
          v16i A = ld_tile128(smem + OFF_A1 + mt * 4096 + 2048, lane);
          pa[mt] = __builtin_amdgcn_wmma_f32_16x16x128_fp8_fp8(A, B, (short)0, pa[mt], false, false);
        }
#pragma unroll
        for (int mt = 0; mt < 2; ++mt) {
#pragma unroll
          for (int r = 0; r < 8; r += 2) {
            int ml = r + (halfsel << 3);
            int m  = mt * 16 + ml;
            out[((size_t)(m0 + m) * pred_steps + d) * 64 + jcol]     = pa[mt][r];
            out[((size_t)(m0 + m + 1) * pred_steps + d) * 64 + jcol] = pa[mt][r + 1];
            unsigned int u = cvt2_fp8(pa[mt][r], pa[mt][r + 1]);
            smem[OFF_A0 + mt * 3072 + a_off(ml, jcol)]     = (unsigned char)(u & 0xFF);
            smem[OFF_A0 + mt * 3072 + a_off(ml + 1, jcol)] = (unsigned char)(u >> 8);
          }
        }
      }
    }
  }
}

extern "C" void kernel_launch(void* const* d_in, const int* in_sizes, int n_in,
                              void* d_out, int out_size, void* d_ws, size_t ws_size,
                              hipStream_t stream) {
  (void)n_in; (void)ws_size;
  const float* x     = (const float*)d_in[0];
  const float* w_ih0 = (const float*)d_in[1];
  const float* w_hh0 = (const float*)d_in[2];
  const float* b_ih0 = (const float*)d_in[3];
  const float* b_hh0 = (const float*)d_in[4];
  const float* w_ih1 = (const float*)d_in[5];
  const float* w_hh1 = (const float*)d_in[6];
  const float* b_ih1 = (const float*)d_in[7];
  const float* b_hh1 = (const float*)d_in[8];
  const float* w_dec = (const float*)d_in[9];
  const float* b_dec = (const float*)d_in[10];
  unsigned char* ws  = (unsigned char*)d_ws;
  float* out = (float*)d_out;

  const int B = in_sizes[0] / (SEQ * INPD);
  const int P = out_size / (B * 64);

  lstm_prep_weights<<<(WEIGHT_BYTES + THREADS - 1) / THREADS, THREADS, 0, stream>>>(
      w_ih0, w_hh0, w_ih1, w_hh1, w_dec, ws);
  lstm_prep_bias<<<5, THREADS, 0, stream>>>(b_ih0, b_hh0, b_ih1, b_hh1, b_dec, ws);
  lstm_main<<<B / MTILE, THREADS, LDS_BYTES, stream>>>(x, ws, out, P);
}
